// self_att_54803782697259
// MI455X (gfx1250) — matrix operand order repaired and checked by reading, not compiled, not run on hardware
//
#include <hip/hip_runtime.h>

typedef __attribute__((ext_vector_type(16))) __bf16 v16bf;
typedef __attribute__((ext_vector_type(8)))  __bf16 v8bf;
typedef __attribute__((ext_vector_type(8)))  float  v8f;

#define BATCH 4
#define NSEQ  4096
#define DDIM  256

// ---------------------------------------------------------------------------
// WMMA helper: D = A(16x32 bf16) * B(32x16 bf16) + C(16x16 f32)
// ---------------------------------------------------------------------------
__device__ __forceinline__ v8f wmma_bf16(v16bf a, v16bf b, v8f c) {
  return __builtin_amdgcn_wmma_f32_16x16x32_bf16(
      /*neg_a=*/false, a, /*neg_b=*/false, b,
      /*c_mod=*/(short)0, c, /*reuse_a=*/false, /*reuse_b=*/false);
}

// A-fragment (16x32 bf16): lane l holds row l%16,
// halves 0..15 = K (sel8 .. sel8+15) contiguous,
// sel8 = (l<16 ? 0 : 16). Two contiguous 16B loads from a row-major row.
__device__ __forceinline__ v16bf load_a_frag(const __bf16* rowptr, int kbase, int lane) {
  const int sel8 = (lane & 16) ? 16 : 0;
  v8bf lo = *(const v8bf*)(rowptr + kbase + sel8);
  v8bf hi = *(const v8bf*)(rowptr + kbase + sel8 + 8);
  return __builtin_shufflevector(lo, hi, 0,1,2,3,4,5,6,7,8,9,10,11,12,13,14,15);
}

__device__ __forceinline__ float sigmoid_f(float x) {
  return __builtin_amdgcn_rcpf(1.0f + __expf(-x));
}

// ---------------------------------------------------------------------------
// Pass 0a: fp32 X -> bf16 row-major Xbf[b][n][d] and bf16 transposed Xt[b][d][n]
// ---------------------------------------------------------------------------
__global__ void cvt_x_kernel(const float* __restrict__ x,
                             __bf16* __restrict__ xbf,
                             __bf16* __restrict__ xt) {
  int idx = blockIdx.x * blockDim.x + threadIdx.x;     // 0 .. 4M-1
  int d = idx & (DDIM - 1);
  int n = (idx >> 8) & (NSEQ - 1);
  int b = idx >> 20;
  float f = x[idx];
  __bf16 h = (__bf16)f;
  xbf[idx] = h;
  xt[(size_t)b * DDIM * NSEQ + (size_t)d * NSEQ + n] = h;
}

// Pass 0b: fp32 W[k][e] -> bf16 Wt[e][k]
__global__ void cvt_w_kernel(const float* __restrict__ w, __bf16* __restrict__ wt) {
  int idx = blockIdx.x * blockDim.x + threadIdx.x;     // 0 .. 65535
  int e = idx & (DDIM - 1);
  int k = idx >> 8;
  wt[(size_t)e * DDIM + k] = (__bf16)w[idx];
}

// ---------------------------------------------------------------------------
// Pass 1: Q = X * W  (bf16 in, f32 acc, bf16 out). One wave per 16x16 tile.
// Rows flattened over B*N (16384), cols over D (256) -> 16384 tiles.
// ---------------------------------------------------------------------------
__global__ __launch_bounds__(256) void qproj_kernel(const __bf16* __restrict__ xbf,
                                                    const __bf16* __restrict__ wt,
                                                    __bf16* __restrict__ qbf) {
  const int lane = threadIdx.x & 31;
  const int wv   = threadIdx.x >> 5;                   // 0..7
  const int tile = blockIdx.x * 8 + wv;                // 0..16383
  const int tm   = tile >> 4;                          // row tile (0..1023)
  const int te   = tile & 15;                          // col tile (0..15)

  const __bf16* arow = xbf + (size_t)(tm * 16 + (lane & 15)) * DDIM;
  const __bf16* brow = wt  + (size_t)(te * 16 + (lane & 15)) * DDIM;
  const int bk = (lane & 16) ? 16 : 0;                 // B-frag K-half select

  v8f acc = {};
#pragma unroll
  for (int kc = 0; kc < 8; ++kc) {
    v16bf a = load_a_frag(arow, kc * 32, lane);
    v16bf b = *(const v16bf*)(brow + kc * 32 + bk);
    acc = wmma_bf16(a, b, acc);
  }

  // C/D layout: lane l, vgpr v -> row v + 8*(l>=16), col l%16
  const int rbase = tm * 16 + ((lane & 16) ? 8 : 0);
  const int col   = te * 16 + (lane & 15);
#pragma unroll
  for (int v = 0; v < 8; ++v)
    qbf[(size_t)(rbase + v) * DDIM + col] = (__bf16)acc[v];
}

// ---------------------------------------------------------------------------
// Pass 2: fused  V = sigmoid(Q * X^T) * X   (streaming over keys, no softmax
// normalization -> pure accumulate, so the key sum is associative).
// One WORKGROUP (4 waves) per 16-query-row tile; wave w handles key blocks
// m = w*32 (mod 128)  -> 4x occupancy vs. one-wave-per-tile, no redundant WMMA.
// Per m-step of 32 keys per wave:
//   S(16x32) via 16 WMMAs, sigmoid, LDS transpose C-layout -> A-layout,
//   V update via 16 WMMAs against Xt.
// Epilogue: 4 partial 16x256 f32 accumulators reduced through LDS.
// ---------------------------------------------------------------------------
__global__ __launch_bounds__(128, 1) void attn_kernel(const __bf16* __restrict__ xbf,
                                                      const __bf16* __restrict__ xt,
                                                      const __bf16* __restrict__ qbf,
                                                      float* __restrict__ out) {
  __shared__ float  racc[4 * 16 * DDIM];               // 64 KB reduction buffer
  __shared__ __bf16 sbuf[4 * 16 * 32];                 // per-wave sigma staging
  const int lane = threadIdx.x & 31;
  const int wv   = threadIdx.x >> 5;                   // 0..3
  const int tile = blockIdx.x;                         // 0..1023
  const int b    = tile >> 8;
  const int n0   = (tile & 255) * 16;

  __bf16* s = sbuf + wv * (16 * 32);
  const __bf16* xb   = xbf + (size_t)b * NSEQ * DDIM;
  const __bf16* xtb  = xt  + (size_t)b * DDIM * NSEQ;
  const __bf16* qrow = qbf + (size_t)(b * NSEQ + n0 + (lane & 15)) * DDIM;

  // Resident Q A-fragments: 16 rows x 256 K  (8 chunks of 16x32)
  v16bf qa[8];
#pragma unroll
  for (int kc = 0; kc < 8; ++kc) qa[kc] = load_a_frag(qrow, kc * 32, lane);

  v8f acc[16];
#pragma unroll
  for (int t = 0; t < 16; ++t) acc[t] = (v8f){};

  const int bk   = (lane & 16) ? 16 : 0;               // B-frag K-half select
  const int col  = lane & 15;
  const int sel8 = (lane & 16) ? 16 : 0;
  const int rb   = (lane & 16) ? 8 : 0;                // C-layout row base

  for (int m0 = wv * 32; m0 < NSEQ; m0 += 128) {
    const __bf16* kr0 = xb + (size_t)(m0 + col) * DDIM;        // keys m0..m0+15
    const __bf16* kr1 = kr0 + (size_t)16 * DDIM;               // keys m0+16..+31
    __builtin_prefetch(kr0 + (size_t)128 * DDIM, 0, 1);        // global_prefetch
    __builtin_prefetch(kr1 + (size_t)128 * DDIM, 0, 1);

    // S = Q * X^T : two 16x16 C tiles (key halves), K-loop over D
    v8f s0 = {}, s1 = {};
#pragma unroll
    for (int kc = 0; kc < 8; ++kc) {
      v16bf b0 = *(const v16bf*)(kr0 + kc * 32 + bk);
      s0 = wmma_bf16(qa[kc], b0, s0);
      v16bf b1 = *(const v16bf*)(kr1 + kc * 32 + bk);
      s1 = wmma_bf16(qa[kc], b1, s1);
    }

    // sigmoid + scatter into LDS as a row-major 16x32 bf16 tile
#pragma unroll
    for (int v = 0; v < 8; ++v) {
      s[(rb + v) * 32 + col]      = (__bf16)sigmoid_f(s0[v]);
      s[(rb + v) * 32 + 16 + col] = (__bf16)sigmoid_f(s1[v]);
    }

    // gather A-fragment of sigma(S): lane row l%16, K = sel8..sel8+15 contiguous
    const __bf16* srow = s + (lane & 15) * 32;
    v8bf lo = *(const v8bf*)(srow + sel8);
    v8bf hi = *(const v8bf*)(srow + sel8 + 8);
    v16bf sa = __builtin_shufflevector(lo, hi, 0,1,2,3,4,5,6,7,8,9,10,11,12,13,14,15);

    // V += sigma(S)(16x32) * X(32x16-per-D-tile), B from transposed Xt
#pragma unroll
    for (int t = 0; t < 16; ++t) {
      v16bf bv = *(const v16bf*)(xtb + (size_t)(t * 16 + col) * NSEQ + m0 + bk);
      acc[t] = wmma_bf16(sa, bv, acc[t]);
    }
  }

  // -------- Reduction epilogue: 4 partial accumulators -> output ----------
  // Each wave dumps its C-layout accumulator into its LDS slab (row-major).
  float* r = racc + wv * (16 * DDIM);
#pragma unroll
  for (int t = 0; t < 16; ++t) {
#pragma unroll
    for (int v = 0; v < 8; ++v)
      r[(rb + v) * DDIM + t * 16 + col] = acc[t][v];
  }
  __syncthreads();

  // Cooperative sum of the 4 slabs; float4 LDS reads + coalesced f32x4 stores.
  float* ob = out + ((size_t)b * NSEQ + n0) * DDIM;
  for (int i = threadIdx.x * 4; i < 16 * DDIM; i += 128 * 4) {
    float4 a0 = *(const float4*)(racc + 0 * 16 * DDIM + i);
    float4 a1 = *(const float4*)(racc + 1 * 16 * DDIM + i);
    float4 a2 = *(const float4*)(racc + 2 * 16 * DDIM + i);
    float4 a3 = *(const float4*)(racc + 3 * 16 * DDIM + i);
    float4 o;
    o.x = a0.x + a1.x + a2.x + a3.x;
    o.y = a0.y + a1.y + a2.y + a3.y;
    o.z = a0.z + a1.z + a2.z + a3.z;
    o.w = a0.w + a1.w + a2.w + a3.w;
    *(float4*)(ob + i) = o;
  }
}

// ---------------------------------------------------------------------------
// Launch
// ---------------------------------------------------------------------------
extern "C" void kernel_launch(void* const* d_in, const int* in_sizes, int n_in,
                              void* d_out, int out_size, void* d_ws, size_t ws_size,
                              hipStream_t stream) {
  const float* x = (const float*)d_in[0];   // [4,4096,256] fp32
  const float* w = (const float*)d_in[1];   // [256,256]    fp32
  float* out = (float*)d_out;               // [4,4096,256] fp32

  const size_t xelems = (size_t)BATCH * NSEQ * DDIM;  // 4M
  char* ws = (char*)d_ws;
  __bf16* xbf = (__bf16*)(ws);                        //  8 MB
  __bf16* xt  = (__bf16*)(ws + xelems * 2);           //  8 MB
  __bf16* qbf = (__bf16*)(ws + xelems * 4);           //  8 MB
  __bf16* wt  = (__bf16*)(ws + xelems * 6);           // 128 KB

  cvt_x_kernel<<<(int)(xelems / 256), 256, 0, stream>>>(x, xbf, xt);
  cvt_w_kernel<<<(DDIM * DDIM) / 256, 256, 0, stream>>>(w, wt);
  // 16384 output tiles, 8 waves (tiles) per 256-thread block
  qproj_kernel<<<(BATCH * NSEQ / 16) * (DDIM / 16) / 8, 256, 0, stream>>>(xbf, wt, qbf);
  // 1024 query tiles, one 4-wave workgroup per tile (keys split across waves)
  attn_kernel<<<BATCH * (NSEQ / 16), 128, 0, stream>>>(xbf, xt, qbf, out);
}
